// DispatchEinsumCombineModel_62878321214344
// MI455X (gfx1250) — matrix operand order repaired and checked by reading, not compiled, not run on hardware
//
#include <hip/hip_runtime.h>
#include <hip/hip_bf16.h>
#include <math.h>

// Problem dims fixed by setup_inputs(): B=2,S=1024 -> T=2048, H=1024, E=8, I=2048.
#define T_TOK  2048
#define HDIM   1024
#define NEXP   8
#define IDIM   2048
#define TWOI   4096
#define GLU_ALPHA 1.702f
#define GLU_LIMIT 7.0f

#define LDS_PITCH 88   // bf16 elements; 176 B row pitch: 16B-aligned, conflict-free

typedef __attribute__((ext_vector_type(16))) __bf16 bf16x16;
typedef __attribute__((ext_vector_type(8)))  __bf16 bf16x8;
typedef __attribute__((ext_vector_type(4)))  __bf16 bf16x4;
typedef __attribute__((ext_vector_type(2)))  __bf16 bf16x2;
typedef __attribute__((ext_vector_type(8)))  float  f32x8;

static __device__ __forceinline__ __bf16 to_bf16(float f) { return (__bf16)f; }

// pack two floats (rows r, r+1 of a transposed tile) into one aligned b32 store
static __device__ __forceinline__ void pack2_store(__bf16* dst, float lo, float hi)
{
  bf16x2 p;
  p[0] = (__bf16)lo;
  p[1] = (__bf16)hi;
  *(bf16x2*)dst = p;
}

static __device__ __forceinline__ bf16x16 cat8(bf16x8 lo, bf16x8 hi)
{
  return __builtin_shufflevector(lo, hi, 0, 1, 2, 3, 4, 5, 6, 7,
                                         8, 9, 10, 11, 12, 13, 14, 15);
}

// ---------------------------------------------------------------------------
// Kernel 1: router.  One wave32 per token; lanes stride H, shfl_xor reduce,
// then lane 0 does softmax over 8 logits + top-2 and bumps expert counts.
// ---------------------------------------------------------------------------
__global__ __launch_bounds__(256) void moe_router(
    const float* __restrict__ X, const float* __restrict__ RW,
    const float* __restrict__ RB, int* __restrict__ topk_idx,
    float* __restrict__ topk_w, int* __restrict__ counts)
{
  const int wave = threadIdx.x >> 5;
  const int lane = threadIdx.x & 31;
  const int t = blockIdx.x * 8 + wave;          // grid is exactly T/8 blocks

  float acc[NEXP];
#pragma unroll
  for (int e = 0; e < NEXP; ++e) acc[e] = 0.0f;

  const float* xr = X + (size_t)t * HDIM;
  for (int h = lane; h < HDIM; h += 32) {
    const float xv = xr[h];
#pragma unroll
    for (int e = 0; e < NEXP; ++e) acc[e] += xv * RW[e * HDIM + h];
  }
#pragma unroll
  for (int e = 0; e < NEXP; ++e) {
#pragma unroll
    for (int off = 16; off > 0; off >>= 1)
      acc[e] += __shfl_xor(acc[e], off, 32);
  }

  if (lane == 0) {
    float lg[NEXP], ex[NEXP];
    float mx = -1e30f;
#pragma unroll
    for (int e = 0; e < NEXP; ++e) { lg[e] = acc[e] + RB[e]; mx = fmaxf(mx, lg[e]); }
    float sum = 0.0f;
#pragma unroll
    for (int e = 0; e < NEXP; ++e) { ex[e] = expf(lg[e] - mx); sum += ex[e]; }
    int i0 = 0;
#pragma unroll
    for (int e = 1; e < NEXP; ++e) if (lg[e] > lg[i0]) i0 = e;   // ties -> lowest idx
    int i1 = (i0 == 0) ? 1 : 0;
#pragma unroll
    for (int e = 0; e < NEXP; ++e) if (e != i0 && lg[e] > lg[i1]) i1 = e;
    const float inv = 1.0f / sum;
    topk_idx[2 * t]     = i0;
    topk_idx[2 * t + 1] = i1;
    topk_w[2 * t]       = ex[i0] * inv;
    topk_w[2 * t + 1]   = ex[i1] * inv;
    atomicAdd(&counts[i0], 1);
    atomicAdd(&counts[i1], 1);
  }
}

// ---------------------------------------------------------------------------
// Kernel 2: exclusive prefix over 8 counts.
// ---------------------------------------------------------------------------
__global__ void moe_offsets(const int* __restrict__ counts, int* __restrict__ offsets)
{
  if (threadIdx.x == 0 && blockIdx.x == 0) {
    int s = 0;
    for (int e = 0; e < NEXP; ++e) { offsets[e] = s; s += counts[e]; }
  }
}

// ---------------------------------------------------------------------------
// Kernel 3: scatter tokens into per-expert buckets; record token->slot map.
// ---------------------------------------------------------------------------
__global__ __launch_bounds__(256) void moe_scatter(
    const int* __restrict__ topk_idx, const int* __restrict__ offsets,
    int* __restrict__ cursors, int* __restrict__ bucket_token,
    int* __restrict__ token_slot)
{
  const int t = blockIdx.x * 256 + threadIdx.x;
  if (t >= T_TOK) return;
#pragma unroll
  for (int k = 0; k < 2; ++k) {
    const int e = topk_idx[2 * t + k];
    const int pos = offsets[e] + atomicAdd(&cursors[e], 1);
    bucket_token[pos] = t;
    token_slot[2 * t + k] = pos;
  }
}

// ---------------------------------------------------------------------------
// Kernel 4: gate_up GEMM + GLU.  Block = 256 thr = 8 waves as 2(M)x4(N).
// Output tile: 32 tokens x 64 intermediate cols.  K_TILE = 64 per barrier
// pair: LDS holds bf16 tiles in fragment-friendly layouts (A row-major,
// B transposed BT[n][k], pitch 88 elem = 176 B) so every lane builds each
// WMMA fragment with exactly two ds_load_b128; 4 wmma per K-iteration.
// ---------------------------------------------------------------------------
__global__ __launch_bounds__(256) void moe_gateup(
    const float* __restrict__ X, const float* __restrict__ GUW,
    const float* __restrict__ GUB, const int* __restrict__ counts,
    const int* __restrict__ offsets, const int* __restrict__ bucket_token,
    __bf16* __restrict__ act)
{
  __shared__ __bf16 Abf[32][LDS_PITCH];   // [m][k]
  __shared__ __bf16 BgT[64][LDS_PITCH];   // [n][k]  (transposed)
  __shared__ __bf16 BuT[64][LDS_PITCH];

  const int e = blockIdx.z;
  const int cnt = counts[e];
  const int mBase = blockIdx.y * 32;
  if (mBase >= cnt) return;                       // block-uniform: EXEC stays all-1s
  const int nBase = blockIdx.x * 64;
  const int off = offsets[e];

  const int tid  = threadIdx.x;
  const int lane = tid & 31, wave = tid >> 5;
  const int wm = wave >> 2, wn = wave & 3;
  const int half = lane >> 4, ln16 = lane & 15;

  // A staging: thread -> (row, two 4-col chunks): 32 rows x 8 chunks (+32)
  const int sRow = tid >> 3;
  const int sC4  = (tid & 7) << 2;
  int sTok = -1;
  if (mBase + sRow < cnt) sTok = bucket_token[off + mBase + sRow];

  // B staging: thread -> (two row pairs, 4-col chunk): 16 pairs (+16) x 16 chunks
  const int bRp = (tid >> 4) << 1;        // rows bRp, bRp+1 (and +32)
  const int bC4 = (tid & 15) << 2;

  const float* W = GUW + (size_t)e * HDIM * TWOI;

  f32x8 cg = {};
  f32x8 cu = {};

  for (int kB = 0; kB < HDIM; kB += 64) {
    __syncthreads();
    // ---- stage A (gathered token rows, bf16, zero-pad past bucket end)
#pragma unroll
    for (int s = 0; s < 2; ++s) {
      const int c = sC4 + 32 * s;
      float4 v = make_float4(0.f, 0.f, 0.f, 0.f);
      if (sTok >= 0) v = *(const float4*)(X + (size_t)sTok * HDIM + kB + c);
      bf16x4 av;
      av[0] = to_bf16(v.x); av[1] = to_bf16(v.y);
      av[2] = to_bf16(v.z); av[3] = to_bf16(v.w);
      *(bf16x4*)&Abf[sRow][c] = av;
    }
    // ---- stage Bg / Bu transposed: two consecutive W rows -> packed b32 stores
#pragma unroll
    for (int s = 0; s < 2; ++s) {
      const int r0 = bRp + 32 * s;
      const float* w0 = W + (size_t)(kB + r0) * TWOI + nBase;
      const float* w1 = w0 + TWOI;
      const float4 g0 = *(const float4*)(w0 + bC4);
      const float4 g1 = *(const float4*)(w1 + bC4);
      const float4 u0 = *(const float4*)(w0 + IDIM + bC4);
      const float4 u1 = *(const float4*)(w1 + IDIM + bC4);
      pack2_store(&BgT[bC4 + 0][r0], g0.x, g1.x);
      pack2_store(&BgT[bC4 + 1][r0], g0.y, g1.y);
      pack2_store(&BgT[bC4 + 2][r0], g0.z, g1.z);
      pack2_store(&BgT[bC4 + 3][r0], g0.w, g1.w);
      pack2_store(&BuT[bC4 + 0][r0], u0.x, u1.x);
      pack2_store(&BuT[bC4 + 1][r0], u0.y, u1.y);
      pack2_store(&BuT[bC4 + 2][r0], u0.z, u1.z);
      pack2_store(&BuT[bC4 + 3][r0], u0.w, u1.w);
    }
    if (kB + 64 < HDIM) {                         // prefetch next K-tile of weights
      const float* wn0 = W + (size_t)(kB + 64 + bRp) * TWOI + nBase;
      __builtin_prefetch(wn0 + bC4, 0, 3);
      __builtin_prefetch(wn0 + IDIM + bC4, 0, 3);
    }
    __syncthreads();

    // ---- two K=32 sub-steps; fragments = two ds_load_b128 per matrix
    const int am = wm * 16 + ln16;
    const int bn = wn * 16 + ln16;
#pragma unroll
    for (int kk = 0; kk < 64; kk += 32) {
      const bf16x16 a = cat8(*(const bf16x8*)&Abf[am][kk + (half << 4)],
                             *(const bf16x8*)&Abf[am][kk + (half << 4) + 8]);
      const bf16x16 bg = cat8(*(const bf16x8*)&BgT[bn][kk + (half << 4)],
                              *(const bf16x8*)&BgT[bn][kk + (half << 4) + 8]);
      const bf16x16 bu = cat8(*(const bf16x8*)&BuT[bn][kk + (half << 4)],
                              *(const bf16x8*)&BuT[bn][kk + (half << 4) + 8]);
      cg = __builtin_amdgcn_wmma_f32_16x16x32_bf16(false, a, false, bg, (short)0, cg, false, false);
      cu = __builtin_amdgcn_wmma_f32_16x16x32_bf16(false, a, false, bu, (short)0, cu, false, false);
    }
  }

  // epilogue: C layout  M = r + 8*half (+16*wm),  N = ln16 (+16*wn)
  const int col = nBase + wn * 16 + ln16;
  const float gb = GUB[(size_t)e * TWOI + col];
  const float ub = GUB[(size_t)e * TWOI + IDIM + col];
#pragma unroll
  for (int r = 0; r < 8; ++r) {
    const int lm = mBase + wm * 16 + r + (half << 3);
    if (lm < cnt) {
      float g = cg[r] + gb;
      float u = cu[r] + ub;
      g = fminf(g, GLU_LIMIT);
      u = fminf(fmaxf(u, -GLU_LIMIT), GLU_LIMIT);
      const float glu = g / (1.0f + expf(-g * GLU_ALPHA));
      act[(size_t)(off + lm) * IDIM + col] = to_bf16((u + 1.0f) * glu);
    }
  }
}

// ---------------------------------------------------------------------------
// Kernel 5: down GEMM.  A fragments load directly from the bf16 activation
// buffer (two contiguous 16B global loads per lane per K=32 sub-step);
// B staged fp32 -> transposed bf16 LDS with K_TILE=64.
// ---------------------------------------------------------------------------
__global__ __launch_bounds__(256) void moe_down(
    const __bf16* __restrict__ act, const float* __restrict__ DW,
    const float* __restrict__ DB, const int* __restrict__ counts,
    const int* __restrict__ offsets, float* __restrict__ pair_out)
{
  __shared__ __bf16 BT[64][LDS_PITCH];    // [n][k] (transposed)

  const int e = blockIdx.z;
  const int cnt = counts[e];
  const int mBase = blockIdx.y * 32;
  if (mBase >= cnt) return;
  const int nBase = blockIdx.x * 64;
  const int off = offsets[e];

  const int tid  = threadIdx.x;
  const int lane = tid & 31, wave = tid >> 5;
  const int wm = wave >> 2, wn = wave & 3;
  const int half = lane >> 4, ln16 = lane & 15;

  const int lmA  = mBase + wm * 16 + ln16;            // this lane's A row
  const int lmAc = (lmA < cnt) ? lmA : (cnt - 1);     // clamp: padding rows unused
  const __bf16* arow = act + (size_t)(off + lmAc) * IDIM;

  const int bRp = (tid >> 4) << 1;
  const int bC4 = (tid & 15) << 2;

  const float* W = DW + (size_t)e * IDIM * HDIM;

  f32x8 c = {};

  for (int kB = 0; kB < IDIM; kB += 64) {
    __syncthreads();
#pragma unroll
    for (int s = 0; s < 2; ++s) {
      const int r0 = bRp + 32 * s;
      const float* w0 = W + (size_t)(kB + r0) * HDIM + nBase;
      const float* w1 = w0 + HDIM;
      const float4 v0 = *(const float4*)(w0 + bC4);
      const float4 v1 = *(const float4*)(w1 + bC4);
      pack2_store(&BT[bC4 + 0][r0], v0.x, v1.x);
      pack2_store(&BT[bC4 + 1][r0], v0.y, v1.y);
      pack2_store(&BT[bC4 + 2][r0], v0.z, v1.z);
      pack2_store(&BT[bC4 + 3][r0], v0.w, v1.w);
    }
    if (kB + 64 < IDIM)
      __builtin_prefetch(W + (size_t)(kB + 64 + bRp) * HDIM + nBase + bC4, 0, 3);
    __syncthreads();

    const int bn = wn * 16 + ln16;
#pragma unroll
    for (int kk = 0; kk < 64; kk += 32) {
      const bf16x16 a = cat8(*(const bf16x8*)(arow + kB + kk + (half << 4)),
                             *(const bf16x8*)(arow + kB + kk + (half << 4) + 8));
      const bf16x16 b = cat8(*(const bf16x8*)&BT[bn][kk + (half << 4)],
                             *(const bf16x8*)&BT[bn][kk + (half << 4) + 8]);
      c = __builtin_amdgcn_wmma_f32_16x16x32_bf16(false, a, false, b, (short)0, c, false, false);
    }
  }

  const int col = nBase + wn * 16 + ln16;
  const float bias = DB[(size_t)e * HDIM + col];
#pragma unroll
  for (int r = 0; r < 8; ++r) {
    const int lm = mBase + wm * 16 + r + (half << 3);
    if (lm < cnt)
      pair_out[(size_t)(off + lm) * HDIM + col] = c[r] + bias;
  }
}

// ---------------------------------------------------------------------------
// Kernel 6: combine  out[t] = w0*pair_out[slot0] + w1*pair_out[slot1].
// ---------------------------------------------------------------------------
__global__ __launch_bounds__(256) void moe_combine(
    const float* __restrict__ pair_out, const int* __restrict__ token_slot,
    const float* __restrict__ topk_w, float* __restrict__ out)
{
  const int idx = blockIdx.x * 256 + threadIdx.x;   // over T*H/4
  const int t  = idx >> 8;                          // H/4 = 256 float4 per token
  const int c4 = (idx & 255) << 2;
  const int   s0 = token_slot[2 * t],   s1 = token_slot[2 * t + 1];
  const float w0 = topk_w[2 * t],       w1 = topk_w[2 * t + 1];
  const float4 a = *(const float4*)(pair_out + (size_t)s0 * HDIM + c4);
  const float4 b = *(const float4*)(pair_out + (size_t)s1 * HDIM + c4);
  float4 r;
  r.x = w0 * a.x + w1 * b.x;
  r.y = w0 * a.y + w1 * b.y;
  r.z = w0 * a.z + w1 * b.z;
  r.w = w0 * a.w + w1 * b.w;
  *(float4*)(out + (size_t)t * HDIM + c4) = r;
}

// ---------------------------------------------------------------------------
extern "C" void kernel_launch(void* const* d_in, const int* in_sizes, int n_in,
                              void* d_out, int out_size, void* d_ws, size_t ws_size,
                              hipStream_t stream)
{
  (void)in_sizes; (void)n_in; (void)out_size; (void)ws_size;

  const float* X   = (const float*)d_in[0];   // [T, H]
  const float* RW  = (const float*)d_in[1];   // [E, H]
  const float* RB  = (const float*)d_in[2];   // [E]
  const float* GUW = (const float*)d_in[3];   // [E, H, 2I]
  const float* GUB = (const float*)d_in[4];   // [E, 2I]
  const float* DW  = (const float*)d_in[5];   // [E, I, H]
  const float* DB  = (const float*)d_in[6];   // [E, H]
  float* out = (float*)d_out;                 // [T, H]

  char* ws = (char*)d_ws;
  int*    counts   = (int*)(ws + 0);                     //  8 ints
  int*    cursors  = (int*)(ws + 32);                    //  8 ints
  int*    offsets  = (int*)(ws + 64);                    //  8 ints
  int*    topk_idx = (int*)(ws + 256);                   //  T*2 ints
  float*  topk_w   = (float*)(ws + 16640);               //  T*2 floats
  int*    tokslot  = (int*)(ws + 33024);                 //  T*2 ints
  int*    bucket   = (int*)(ws + 49408);                 //  2T ints
  __bf16* act      = (__bf16*)(ws + 65792);              //  [2T, I] bf16  (16 MB)
  float*  pair_out = (float*)(ws + 65792 + (size_t)2 * T_TOK * IDIM * 2); // [2T, H] f32 (16 MB)

  hipMemsetAsync(d_ws, 0, 96, stream);  // counts + cursors (+offsets) = 0

  moe_router <<<T_TOK / 8, 256, 0, stream>>>(X, RW, RB, topk_idx, topk_w, counts);
  moe_offsets<<<1, 32, 0, stream>>>(counts, offsets);
  moe_scatter<<<(T_TOK + 255) / 256, 256, 0, stream>>>(topk_idx, offsets, cursors,
                                                       bucket, tokslot);
  moe_gateup <<<dim3(IDIM / 64, 64, NEXP), 256, 0, stream>>>(X, GUW, GUB, counts,
                                                             offsets, bucket, act);
  moe_down   <<<dim3(HDIM / 64, 64, NEXP), 256, 0, stream>>>(act, DW, DB, counts,
                                                             offsets, pair_out);
  moe_combine<<<(T_TOK * HDIM / 4) / 256, 256, 0, stream>>>(pair_out, tokslot,
                                                            topk_w, out);
}
